// AttentionPattern_83820581749443
// MI455X (gfx1250) — matrix operand order repaired and checked by reading, not compiled, not run on hardware
//
#include <hip/hip_runtime.h>

// ---------------------------------------------------------------------------
// Attention (B=4, S=2048, D=1024), f16 WMMA pipeline:
//   1) Qh = f16(x@Wq^T + bq), Kh = ..., Vt = transpose(f16(x@Wv^T + bv))
//   2) Ph = f16( (Qh@Kh^T) * (1/sqrt(D)) * gaussian_mask[key] )
//   3) Ph = softmax_rows(Ph)
//   4) Oh = f16( Ph @ V )            (NT against Vt)
//   5) out = Oh@Wo^T + bo  (f32)
// GEMM: 128x128 tiles, 8 wave32 waves (64x32 each), double-buffered LDS,
// register-staged software pipeline, v_wmma_f32_16x16x32_f16, f32 accum.
// ---------------------------------------------------------------------------

typedef _Float16 h4   __attribute__((ext_vector_type(4)));
typedef _Float16 h8   __attribute__((ext_vector_type(8)));
typedef _Float16 v16h __attribute__((ext_vector_type(16)));
typedef float    v8f  __attribute__((ext_vector_type(8)));

#define TILE  128
#define KSTEP 64
#define LDSS  72   // LDS row stride in halfs (64 + 8 pad => 144B rows, conflict-free 16B lanes)

enum { EPI_BIAS_F16 = 0, EPI_SCALEMASK_F16 = 1, EPI_BIAS_F32 = 2, EPI_BIAS_F16_T = 3 };

__device__ __forceinline__ v16h make_v16(h8 lo, h8 hi) {
  return __builtin_shufflevector(lo, hi, 0,1,2,3,4,5,6,7,8,9,10,11,12,13,14,15);
}

template<bool A_F32, bool B_F32, int EPI>
__global__ __launch_bounds__(256)
void gemm_nt_wmma(const void* __restrict__ Ap, const void* __restrict__ Bp,
                  const float* __restrict__ bias, void* __restrict__ Cp,
                  int K, int lda, int ldb, int ldc,
                  long sA, long sB, long sC, float scale, int ldct)
{
  __shared__ _Float16 As[2][TILE][LDSS];
  __shared__ _Float16 Bs[2][TILE][LDSS];

  const int tid  = threadIdx.x;
  const int lane = tid & 31;
  const int w    = tid >> 5;          // 8 waves
  const int wm   = (w & 1) * 64;      // 2 x 4 wave grid: 64x32 C-subtile per wave
  const int wn   = (w >> 1) * 32;
  const int tM   = blockIdx.y * TILE;
  const int tN   = blockIdx.x * TILE;

  const char* Ab = (const char*)Ap + (size_t)blockIdx.z * (size_t)sA * (A_F32 ? 4u : 2u);
  const char* Bb = (const char*)Bp + (size_t)blockIdx.z * (size_t)sB * (B_F32 ? 4u : 2u);

  // ---- branch-free staging coordinates (compile-time trip counts) ----
  // f32 tiles: 128 rows x 16 float4-chunks; thread owns chunk (tid&15) of rows (tid>>4)+16*it
  // f16 tiles: 128 rows x 8 h8-chunks;     thread owns chunk (tid&7)  of rows (tid>>3)+32*it
  const int r4 = tid >> 4, c4 = (tid & 15) * 4;
  const int r8 = tid >> 3, c8 = (tid & 7) * 8;

  float4 rAf[8]; h8 rAh[4];
  float4 rBf[8]; h8 rBh[4];

  auto loadA = [&](int kt) {
    if (A_F32) {
      const float* A = (const float*)Ab + (size_t)(tM + r4) * lda + c4 + kt;
      #pragma unroll
      for (int it = 0; it < 8; ++it) rAf[it] = *(const float4*)(A + (size_t)it * 16 * lda);
    } else {
      const _Float16* A = (const _Float16*)Ab + (size_t)(tM + r8) * lda + c8 + kt;
      #pragma unroll
      for (int it = 0; it < 4; ++it) rAh[it] = *(const h8*)(A + (size_t)it * 32 * lda);
    }
  };
  auto loadB = [&](int kt) {
    if (B_F32) {
      const float* Bm = (const float*)Bb + (size_t)(tN + r4) * ldb + c4 + kt;
      #pragma unroll
      for (int it = 0; it < 8; ++it) rBf[it] = *(const float4*)(Bm + (size_t)it * 16 * ldb);
    } else {
      const _Float16* Bm = (const _Float16*)Bb + (size_t)(tN + r8) * ldb + c8 + kt;
      #pragma unroll
      for (int it = 0; it < 4; ++it) rBh[it] = *(const h8*)(Bm + (size_t)it * 32 * ldb);
    }
  };
  auto storeA = [&](int buf) {
    if (A_F32) {
      #pragma unroll
      for (int it = 0; it < 8; ++it) {
        h4 hv = { (_Float16)rAf[it].x, (_Float16)rAf[it].y,
                  (_Float16)rAf[it].z, (_Float16)rAf[it].w };
        *(h4*)&As[buf][r4 + it * 16][c4] = hv;
      }
    } else {
      #pragma unroll
      for (int it = 0; it < 4; ++it) *(h8*)&As[buf][r8 + it * 32][c8] = rAh[it];
    }
  };
  auto storeB = [&](int buf) {
    if (B_F32) {
      #pragma unroll
      for (int it = 0; it < 8; ++it) {
        h4 hv = { (_Float16)rBf[it].x, (_Float16)rBf[it].y,
                  (_Float16)rBf[it].z, (_Float16)rBf[it].w };
        *(h4*)&Bs[buf][r4 + it * 16][c4] = hv;
      }
    } else {
      #pragma unroll
      for (int it = 0; it < 4; ++it) *(h8*)&Bs[buf][r8 + it * 32][c8] = rBh[it];
    }
  };

  v8f acc[4][2] = {};
  const int r  = lane & 15;
  const int hi = lane >> 4;

  // ---- software pipeline: global->regs (next) overlaps LDS->WMMA (current) ----
  loadA(0); loadB(0);
  int buf = 0;
  for (int kt = 0; kt < K; kt += KSTEP) {
    storeA(buf); storeB(buf);            // waits staged loads, converts, ds_store
    __syncthreads();                     // one barrier per K-step (double buffered)
    if (kt + KSTEP < K) { loadA(kt + KSTEP); loadB(kt + KSTEP); }  // issue early

    #pragma unroll
    for (int kk = 0; kk < KSTEP; kk += 32) {
      v16h a[4], b[2];
      #pragma unroll
      for (int i = 0; i < 4; ++i) {
        const _Float16* p = &As[buf][wm + i * 16 + r][kk + hi * 8];
        a[i] = make_v16(*(const h8*)p, *(const h8*)(p + 16));
      }
      #pragma unroll
      for (int j = 0; j < 2; ++j) {
        const _Float16* p = &Bs[buf][wn + j * 16 + r][kk + hi * 8];
        b[j] = make_v16(*(const h8*)p, *(const h8*)(p + 16));
      }
      #pragma unroll
      for (int i = 0; i < 4; ++i)
        #pragma unroll
        for (int j = 0; j < 2; ++j)
          acc[i][j] = __builtin_amdgcn_wmma_f32_16x16x32_f16(
              false, a[i], false, b[j], (short)0, acc[i][j], false, false);
    }
    buf ^= 1;
  }

  // ---- epilogue: C layout lane = N col, VGPR e = row e (+8 for hi lanes) ----
  char* Cb = (char*)Cp + (size_t)blockIdx.z * (size_t)sC * (EPI == EPI_BIAS_F32 ? 4u : 2u);
  const int rowb = tM + wm + hi * 8;
  const int col0 = tN + wn + r;
  #pragma unroll
  for (int j = 0; j < 2; ++j) {
    const int c = col0 + j * 16;
    float bv = 0.0f, mv = 1.0f;
    if (EPI == EPI_SCALEMASK_F16) {
      const float nk = (float)(gridDim.x * TILE);          // number of keys (S)
      const float d  = ((float)c - 0.5f * nk) / (0.25f * nk);
      mv = scale * __expf(-0.5f * d * d);                  // (1/sqrt(D)) * gaussian(key)
    } else if (bias) {
      bv = bias[c];
    }
    #pragma unroll
    for (int i = 0; i < 4; ++i) {
      #pragma unroll
      for (int e = 0; e < 8; ++e) {
        const int rr  = rowb + i * 16 + e;
        const float v = acc[i][j][e];
        if (EPI == EPI_BIAS_F16) {
          ((_Float16*)Cb)[(size_t)rr * ldc + c] = (_Float16)(v + bv);
        } else if (EPI == EPI_SCALEMASK_F16) {
          ((_Float16*)Cb)[(size_t)rr * ldc + c] = (_Float16)(v * mv);
        } else if (EPI == EPI_BIAS_F32) {
          ((float*)Cb)[(size_t)rr * ldc + c] = v + bv;
        } else { // EPI_BIAS_F16_T : C[c][rr], row stride ldct (for Vt)
          ((_Float16*)Cb)[(size_t)c * ldct + rr] = (_Float16)(v + bv);
        }
      }
    }
  }
}

// Row softmax over 2048 f16 entries, in place. One 256-thread block per row.
__global__ __launch_bounds__(256)
void softmax_rows2048(_Float16* __restrict__ P)
{
  __shared__ float red[256];
  const int tid = threadIdx.x;
  _Float16* row = P + (size_t)blockIdx.x * 2048;
  float v[8];
  #pragma unroll
  for (int i = 0; i < 8; ++i) v[i] = (float)row[tid + (i << 8)];
  float m = v[0];
  #pragma unroll
  for (int i = 1; i < 8; ++i) m = fmaxf(m, v[i]);
  red[tid] = m;
  __syncthreads();
  for (int s = 128; s > 0; s >>= 1) {
    if (tid < s) red[tid] = fmaxf(red[tid], red[tid + s]);
    __syncthreads();
  }
  m = red[0];
  __syncthreads();
  float sum = 0.0f;
  #pragma unroll
  for (int i = 0; i < 8; ++i) { v[i] = __expf(v[i] - m); sum += v[i]; }
  red[tid] = sum;
  __syncthreads();
  for (int s = 128; s > 0; s >>= 1) {
    if (tid < s) red[tid] += red[tid + s];
    __syncthreads();
  }
  const float inv = 1.0f / red[0];
  #pragma unroll
  for (int i = 0; i < 8; ++i) row[tid + (i << 8)] = (_Float16)(v[i] * inv);
}

extern "C" void kernel_launch(void* const* d_in, const int* in_sizes, int n_in,
                              void* d_out, int out_size, void* d_ws, size_t ws_size,
                              hipStream_t stream)
{
  const float* x  = (const float*)d_in[0];
  const float* Wq = (const float*)d_in[1];
  const float* bq = (const float*)d_in[2];
  const float* Wk = (const float*)d_in[3];
  const float* bk = (const float*)d_in[4];
  const float* Wv = (const float*)d_in[5];
  const float* bv = (const float*)d_in[6];
  const float* Wo = (const float*)d_in[7];
  const float* bo = (const float*)d_in[8];

  const int B = 4, S = 2048, D = 1024;
  const size_t M = (size_t)B * S;   // 8192

  // workspace layout (f16 elements): Qh | Kh | Vt | Ph | Oh  (~100.7 MB)
  const size_t need = (4 * M * D + (size_t)B * S * S) * sizeof(_Float16);
  if (ws_size < need) return;

  _Float16* Qh = (_Float16*)d_ws;          // [M][D]
  _Float16* Kh = Qh + M * D;               // [M][D]
  _Float16* Vt = Kh + M * D;               // [D][M]   (V transposed; batch = column block)
  _Float16* Ph = Vt + M * D;               // [B][S][S]
  _Float16* Oh = Ph + (size_t)B * S * S;   // [M][D]

  dim3 blk(256);

  // 1) QKV projections (f32 in -> f16 out). V stored transposed.
  dim3 gQKV(D / TILE, (unsigned)(M / TILE), 1);
  gemm_nt_wmma<true, true, EPI_BIAS_F16><<<gQKV, blk, 0, stream>>>(
      x, Wq, bq, Qh, D, D, D, D, 0, 0, 0, 1.0f, 0);
  gemm_nt_wmma<true, true, EPI_BIAS_F16><<<gQKV, blk, 0, stream>>>(
      x, Wk, bk, Kh, D, D, D, D, 0, 0, 0, 1.0f, 0);
  gemm_nt_wmma<true, true, EPI_BIAS_F16_T><<<gQKV, blk, 0, stream>>>(
      x, Wv, bv, Vt, D, D, D, 0, 0, 0, 0, 1.0f, (int)M);

  // 2) scores = (Q@K^T) * (1/sqrt(D)) * gaussian_mask[key], batched over B
  dim3 gS(S / TILE, S / TILE, B);
  gemm_nt_wmma<false, false, EPI_SCALEMASK_F16><<<gS, blk, 0, stream>>>(
      Qh, Kh, nullptr, Ph, D, D, D, S,
      (long)S * D, (long)S * D, (long)S * S, 0.03125f /* 1/sqrt(1024) */, 0);

  // 3) row softmax (in place)
  softmax_rows2048<<<dim3((unsigned)M), blk, 0, stream>>>(Ph);

  // 4) O = P @ V   (NT against Vt: B rows = output d-columns, K = keys)
  dim3 gPV(D / TILE, S / TILE, B);
  gemm_nt_wmma<false, false, EPI_BIAS_F16><<<gPV, blk, 0, stream>>>(
      Ph, Vt, nullptr, Oh, S, S, (int)M, D,
      (long)S * S, (long)S, (long)S * D, 1.0f, 0);

  // 5) out = O @ Wo^T + bo   (f32 output)
  dim3 gO(D / TILE, (unsigned)(M / TILE), 1);
  gemm_nt_wmma<false, true, EPI_BIAS_F32><<<gO, blk, 0, stream>>>(
      Oh, Wo, bo, (float*)d_out, D, D, D, D, 0, 0, 0, 1.0f, 0);
}
